// Conv2d_19301583028400
// MI455X (gfx1250) — matrix operand order repaired and checked by reading, not compiled, not run on hardware
//
#include <hip/hip_runtime.h>

// ---- CDNA5 WMMA types -------------------------------------------------------
typedef __attribute__((ext_vector_type(16))) __bf16         v16bf;
typedef __attribute__((ext_vector_type(8)))  float          v8f;
typedef __attribute__((ext_vector_type(8)))  unsigned short v8us;

#define CIN   128
#define COUT  256
#define HHW   3136   // 56*56
#define WID   56
#define HP    58     // padded spatial (1-px halo)
#define KTOT  1152   // 9 taps * 128 channels, tap-major: k = tap*128 + ci
#define KC    32
#define NCH   (KTOT / KC)   // 36 chunks
#define KPAD  40     // LDS row stride (elems): 80B, 16B-aligned, bank-friendly
#define MT    128
#define NT    64

union FragU { uint4 q[2]; v16bf v; };

// A-matrix 16x32 bf16 fragment (ISA 7.12.2): lane holds row M=lane&15,
// K elems { khalf*8..+7 } then { 16+khalf*8..+7 }.
__device__ __forceinline__ v16bf load_frag_a(const unsigned short* row, int khalf) {
  FragU u;
  u.q[0] = *(const uint4*)(row + khalf * 8);
  u.q[1] = *(const uint4*)(row + 16 + khalf * 8);
  return u.v;
}
// B-matrix 32x16 bf16 fragment: lane holds column N=lane&15,
// K elems { khalf*8..+7 } then { 16+khalf*8..+7 } (same as A).
__device__ __forceinline__ v16bf load_frag_b(const unsigned short* row, int khalf) {
  FragU u;
  u.q[0] = *(const uint4*)(row + khalf * 8);
  u.q[1] = *(const uint4*)(row + 16 + khalf * 8);
  return u.v;
}

// split f32 -> bf16 hi + bf16 lo (truncation split; dropped lo*lo ~2^-16 rel)
__device__ __forceinline__ void split_bf16(float v, unsigned short& hb, unsigned short& lb) {
  unsigned u = __float_as_uint(v);
  hb = (unsigned short)(u >> 16);
  float hf = __uint_as_float((unsigned)hb << 16);
  lb = (unsigned short)(__float_as_uint(v - hf) >> 16);
}

// ===== prepass 1: weights -> split bf16, layout [co][tap][ci] ================
__global__ __launch_bounds__(256)
void prep_weights(const float* __restrict__ F,
                  unsigned short* __restrict__ Whi,
                  unsigned short* __restrict__ Wlo) {
  int g   = blockIdx.x * 256 + threadIdx.x;   // < 256*9*128
  int ci  = g & (CIN - 1);
  int rst = g >> 7;
  int tap = rst % 9;
  int co  = rst / 9;
  float v = F[(size_t)co * KTOT + ci * 9 + tap];
  unsigned short hb, lb; split_bf16(v, hb, lb);
  size_t o = ((size_t)co * 9 + tap) * CIN + ci;
  Whi[o] = hb; Wlo[o] = lb;
}

// ===== prepass 2: input -> split bf16, transposed [b][h+1][w+1][ci], halo=0 ==
__global__ __launch_bounds__(256)
void prep_input(const float* __restrict__ X,
                unsigned short* __restrict__ Xhi,
                unsigned short* __restrict__ Xlo) {
  int g   = blockIdx.x * 256 + threadIdx.x;   // < 32*3136*128
  int ci  = g & (CIN - 1);
  int rst = g >> 7;                           // b*HHW + hw
  int b   = rst / HHW;
  int hw  = rst - b * HHW;
  int h   = hw / WID;
  int w   = hw - h * WID;
  float v = X[((size_t)b * CIN + ci) * HHW + hw];
  unsigned short hb, lb; split_bf16(v, hb, lb);
  size_t o = (((size_t)b * HP + (h + 1)) * HP + (w + 1)) * CIN + ci;
  Xhi[o] = hb; Xlo[o] = lb;
}

// ===== main: implicit GEMM, pre-split bf16, ping-pong LDS, unrolled K ========
__global__ __launch_bounds__(256)
void conv3x3_wmma_main(const unsigned short* __restrict__ Xhi,
                       const unsigned short* __restrict__ Xlo,
                       const unsigned short* __restrict__ Whi,
                       const unsigned short* __restrict__ Wlo,
                       const float* __restrict__ Bias,
                       float* __restrict__ Out) {
  __shared__ __align__(16) unsigned short Ah[2][MT * KPAD];
  __shared__ __align__(16) unsigned short Al[2][MT * KPAD];
  __shared__ __align__(16) unsigned short Bh[2][NT * KPAD];
  __shared__ __align__(16) unsigned short Bl[2][NT * KPAD];

  const int t     = threadIdx.x;
  const int lane  = t & 31;
  const int wave  = t >> 5;      // 0..7 : M-tile index
  const int row16 = lane & 15;
  const int khalf = lane >> 4;

  const int m_base = blockIdx.y * MT;
  const int n_base = blockIdx.x * NT;

  const int arow = t >> 1;            // A fill: 2 threads/row, 16 elems
  const int fcA  = (t & 1) * 16;
  const int brow = t >> 2;            // B fill: 4 threads/row, 8 elems
  const int fcB  = (t & 3) * 8;

  const int ngr = n_base + brow;
  const int bb  = ngr / HHW;
  const int hwr = ngr - bb * HHW;
  const int hh  = hwr / WID;
  const int ww  = hwr - hh * WID;

  // loop-invariant per-thread base pointers; per-chunk deltas become immediates
  const unsigned short* WhB = Whi + (size_t)(m_base + arow) * 9 * CIN + fcA;
  const unsigned short* WlB = Wlo + (size_t)(m_base + arow) * 9 * CIN + fcA;
  const size_t xinv = (((size_t)bb * HP + hh) * HP + ww) * CIN + fcB; // tap(0,0)
  const unsigned short* XhB = Xhi + xinv;
  const unsigned short* XlB = Xlo + xinv;

  v8f a0 = {0,0,0,0,0,0,0,0}, a1 = a0, a2 = a0, a3 = a0;
  uint4 avh0, avh1, avl0, avl1, bvh, bvl;   // pipeline registers

  auto load_chunk = [&](int kc) {           // kc compile-time after unroll
    const int tap = kc >> 2;
    const int ci0 = (kc & 3) * 32;
    const int kh  = tap / 3;
    const int kw  = tap - kh * 3;
    const int da  = tap * CIN + ci0;              // <= 1120 elems
    const int dx  = (kh * HP + kw) * CIN + ci0;   // <= 15,200 elems
    avh0 = *(const uint4*)(WhB + da);  avh1 = *(const uint4*)(WhB + da + 8);
    avl0 = *(const uint4*)(WlB + da);  avl1 = *(const uint4*)(WlB + da + 8);
    bvh  = *(const uint4*)(XhB + dx);
    bvl  = *(const uint4*)(XlB + dx);
  };
  auto store_chunk = [&](int p) {
    *(uint4*)(&Ah[p][arow * KPAD + fcA])     = avh0;
    *(uint4*)(&Ah[p][arow * KPAD + fcA + 8]) = avh1;
    *(uint4*)(&Al[p][arow * KPAD + fcA])     = avl0;
    *(uint4*)(&Al[p][arow * KPAD + fcA + 8]) = avl1;
    *(uint4*)(&Bh[p][brow * KPAD + fcB])     = bvh;
    *(uint4*)(&Bl[p][brow * KPAD + fcB])     = bvl;
  };

  // prologue: buffer 0 primed, chunk-1 data in flight
  load_chunk(0);
  store_chunk(0);
  load_chunk(1);
  __syncthreads();

#pragma unroll
  for (int kc = 0; kc < NCH; ++kc) {
    const int p = kc & 1;
    if (kc + 1 < NCH) store_chunk(1 - p);   // other buffer: overlaps compute
    if (kc + 2 < NCH) load_chunk(kc + 2);   // global loads in flight

    // ---- 12 WMMAs per wave from buffer p ----
    const unsigned short* arh = &Ah[p][(wave * 16 + row16) * KPAD];
    const unsigned short* arl = &Al[p][(wave * 16 + row16) * KPAD];
    v16bf ahi = load_frag_a(arh, khalf);
    v16bf alo = load_frag_a(arl, khalf);
#pragma unroll
    for (int nt = 0; nt < 4; ++nt) {
      const int nr = (nt * 16 + row16) * KPAD;
      v16bf bh = load_frag_b(&Bh[p][nr], khalf);
      v16bf bl = load_frag_b(&Bl[p][nr], khalf);
      v8f& c = (nt == 0) ? a0 : (nt == 1) ? a1 : (nt == 2) ? a2 : a3;
      c = __builtin_amdgcn_wmma_f32_16x16x32_bf16(false, ahi, false, bl, (short)0, c, false, false);
      c = __builtin_amdgcn_wmma_f32_16x16x32_bf16(false, alo, false, bh, (short)0, c, false, false);
      c = __builtin_amdgcn_wmma_f32_16x16x32_bf16(false, ahi, false, bh, (short)0, c, false, false);
    }
    if (kc + 1 < NCH) __syncthreads();  // next buffer ready; readers of p done
  }

  // ---- epilogue ----
  const int ob  = n_base / HHW;
  const int hw0 = n_base - ob * HHW;
  float* opb = Out + (size_t)ob * COUT * HHW + hw0;
#pragma unroll
  for (int nt = 0; nt < 4; ++nt) {
    v8f c = (nt == 0) ? a0 : (nt == 1) ? a1 : (nt == 2) ? a2 : a3;
    const int ncol = nt * 16 + row16;
#pragma unroll
    for (int r = 0; r < 8; ++r) {
      const int co = m_base + wave * 16 + khalf * 8 + r;
      opb[(size_t)co * HHW + ncol] = c[r] + Bias[co];
    }
  }
}

// ===== fallback (round-2 kernel): in-loop conversion, no workspace ===========
__global__ __launch_bounds__(256)
void conv3x3_fallback(const float* __restrict__ X, const float* __restrict__ F,
                      const float* __restrict__ Bias, float* __restrict__ Out) {
  __shared__ __align__(16) unsigned short Ahi[MT * KPAD];
  __shared__ __align__(16) unsigned short Alo[MT * KPAD];
  __shared__ __align__(16) unsigned short Bhi[NT * KPAD];
  __shared__ __align__(16) unsigned short Blo[NT * KPAD];

  const int t = threadIdx.x, lane = t & 31, wave = t >> 5;
  const int row16 = lane & 15, khalf = lane >> 4;
  const int m_base = blockIdx.y * MT, n_base = blockIdx.x * NT;
  const int arow = t >> 1, fcA = (t & 1) * 16;
  const int brow = t >> 2, fcB = (t & 3) * 8;
  const int ngr = n_base + brow;
  const int bb = ngr / HHW, hwr = ngr - bb * HHW;
  const int hh = hwr / WID, ww = hwr - hh * WID;
  const float* Abase = F + (size_t)(m_base + arow) * KTOT;
  const float* Xb    = X + (size_t)bb * CIN * HHW;

  v8f a0 = {0,0,0,0,0,0,0,0}, a1 = a0, a2 = a0, a3 = a0;
  float av[16]; float bv[8];

  auto load_chunk = [&](int kc) {
    const int tap = kc >> 2, ci0 = (kc & 3) * 32;
    const int kh = tap / 3, kw = tap - kh * 3;
    const float* ap = Abase + (size_t)(ci0 + fcA) * 9 + tap;
#pragma unroll
    for (int j = 0; j < 16; ++j) av[j] = ap[j * 9];
    const int ih = hh + kh - 1, iw = ww + kw - 1;
    if ((unsigned)ih < 56u && (unsigned)iw < 56u) {
      const float* bp = Xb + (size_t)(ci0 + fcB) * HHW + ih * WID + iw;
#pragma unroll
      for (int j = 0; j < 8; ++j) bv[j] = bp[(size_t)j * HHW];
    } else {
#pragma unroll
      for (int j = 0; j < 8; ++j) bv[j] = 0.0f;
    }
  };

  load_chunk(0);
  for (int kc = 0; kc < NCH; ++kc) {
    __syncthreads();
    {
      v8us h0, l0, h1, l1;
#pragma unroll
      for (int j = 0; j < 8; ++j) {
        unsigned short hb, lb;
        split_bf16(av[j], hb, lb);     h0[j] = hb; l0[j] = lb;
        split_bf16(av[8 + j], hb, lb); h1[j] = hb; l1[j] = lb;
      }
      *(v8us*)(&Ahi[arow * KPAD + fcA]) = h0;  *(v8us*)(&Ahi[arow * KPAD + fcA + 8]) = h1;
      *(v8us*)(&Alo[arow * KPAD + fcA]) = l0;  *(v8us*)(&Alo[arow * KPAD + fcA + 8]) = l1;
      v8us hb8, lb8;
#pragma unroll
      for (int j = 0; j < 8; ++j) {
        unsigned short hb, lb; split_bf16(bv[j], hb, lb); hb8[j] = hb; lb8[j] = lb;
      }
      *(v8us*)(&Bhi[brow * KPAD + fcB]) = hb8; *(v8us*)(&Blo[brow * KPAD + fcB]) = lb8;
    }
    __syncthreads();
    if (kc + 1 < NCH) load_chunk(kc + 1);
    const unsigned short* arh = Ahi + (wave * 16 + row16) * KPAD;
    const unsigned short* arl = Alo + (wave * 16 + row16) * KPAD;
    v16bf ahi = load_frag_a(arh, khalf);
    v16bf alo = load_frag_a(arl, khalf);
#pragma unroll
    for (int nt = 0; nt < 4; ++nt) {
      const int nr = (nt * 16 + row16) * KPAD;
      v16bf bh = load_frag_b(Bhi + nr, khalf);
      v16bf bl = load_frag_b(Blo + nr, khalf);
      v8f& c = (nt == 0) ? a0 : (nt == 1) ? a1 : (nt == 2) ? a2 : a3;
      c = __builtin_amdgcn_wmma_f32_16x16x32_bf16(false, ahi, false, bl, (short)0, c, false, false);
      c = __builtin_amdgcn_wmma_f32_16x16x32_bf16(false, alo, false, bh, (short)0, c, false, false);
      c = __builtin_amdgcn_wmma_f32_16x16x32_bf16(false, ahi, false, bh, (short)0, c, false, false);
    }
  }
  const int ob = n_base / HHW, hw0 = n_base - ob * HHW;
  float* opb = Out + (size_t)ob * COUT * HHW + hw0;
#pragma unroll
  for (int nt = 0; nt < 4; ++nt) {
    v8f c = (nt == 0) ? a0 : (nt == 1) ? a1 : (nt == 2) ? a2 : a3;
    const int ncol = nt * 16 + row16;
#pragma unroll
    for (int r = 0; r < 8; ++r) {
      const int co = m_base + wave * 16 + khalf * 8 + r;
      opb[(size_t)co * HHW + ncol] = c[r] + Bias[co];
    }
  }
}

extern "C" void kernel_launch(void* const* d_in, const int* in_sizes, int n_in,
                              void* d_out, int out_size, void* d_ws, size_t ws_size,
                              hipStream_t stream) {
  (void)in_sizes; (void)n_in; (void)out_size;
  const float* x    = (const float*)d_in[0];
  const float* filt = (const float*)d_in[1];
  const float* bias = (const float*)d_in[2];
  float* out        = (float*)d_out;

  const size_t W_E = (size_t)COUT * 9 * CIN;        // 294,912 elems
  const size_t X_E = (size_t)32 * HP * HP * CIN;    // 13,778,944 elems
  const size_t need = 2 * 2 * (W_E + X_E);          // ~53.7 MiB

  if (d_ws && ws_size >= need) {
    unsigned short* whi = (unsigned short*)d_ws;
    unsigned short* wlo = whi + W_E;
    unsigned short* xhi = wlo + W_E;
    unsigned short* xlo = xhi + X_E;
    // zero the padded inputs (establishes the conv halo); graph-capturable
    hipMemsetAsync(xhi, 0, 2 * 2 * X_E, stream);
    prep_weights<<<(unsigned)(W_E / 256), 256, 0, stream>>>(filt, whi, wlo);
    prep_input  <<<(unsigned)(((size_t)32 * HHW * CIN) / 256), 256, 0, stream>>>(x, xhi, xlo);
    conv3x3_wmma_main<<<dim3(1568, 2), 256, 0, stream>>>(xhi, xlo, whi, wlo, bias, out);
  } else {
    conv3x3_fallback<<<dim3(1568, 2), 256, 0, stream>>>(x, filt, bias, out);
  }
}
